// MultiSpeciesMLP_76544907149626
// MI455X (gfx1250) — matrix operand order repaired and checked by reading, not compiled, not run on hardware
//
#include <hip/hip_runtime.h>
#include <hip/hip_bf16.h>

typedef _Float16 f16;
typedef __attribute__((ext_vector_type(16))) _Float16 v16h;
typedef __attribute__((ext_vector_type(8)))  _Float16 v8h;
typedef __attribute__((ext_vector_type(8)))  float    v8f;

#define N_ATOMS 65536
#define N_IN    256
#define N_HID   512
#define N_SP    4
#define MT      32      // atoms (M rows) per block: B-fragments reused 2x

// Workspace layout (bytes)
#define CNT_OFF 0
#define IDX_OFF 256
#define W1H_OFF (IDX_OFF + N_SP * N_ATOMS * 4)          // 1,048,832
#define W2H_OFF (W1H_OFF + N_SP * N_HID * N_IN * 2)     // + 1 MB
// total ~4.2 MB

// LDS row strides (f16 elements), padded to dodge bank conflicts
#define XS (N_IN + 8)    // 264
#define HS (N_HID + 8)   // 520

// ---------------------------------------------------------------------------
// Hardware tanh (CDNA5 V_TANH_F32, a TRANS32 op) instead of libm expansion.
__device__ inline float fast_tanh(float v) {
#if __has_builtin(__builtin_amdgcn_tanhf)
  return __builtin_amdgcn_tanhf(v);
#elif __has_builtin(__builtin_amdgcn_tanh_f32)
  return __builtin_amdgcn_tanh_f32(v);
#else
  float r;
  asm("v_tanh_f32 %0, %1" : "=v"(r) : "v"(v));
  return r;
#endif
}

// ---------------------------------------------------------------------------
// A-fragment (16x32 f16, M x K) from LDS. Per ISA table:
//   lanes 0-15 : M=lane,    elems 0-7 = K 0..7,  elems 8-15 = K 16..23
//   lanes16-31 : M=lane-16, elems 0-7 = K 8..15, elems 8-15 = K 24..31
__device__ inline v16h load_afrag(const f16* __restrict__ sA, int stride,
                                  int rowbase, int lane, int kstep) {
  const int row = rowbase + (lane & 15);
  const int kb0 = kstep * 32 + ((lane < 16) ? 0 : 8);
  const v8h* p = reinterpret_cast<const v8h*>(sA + row * stride + kb0);
  v8h lo = p[0];
  v8h hi = p[2];  // +16 f16 elements
  return __builtin_shufflevector(lo, hi, 0,1,2,3,4,5,6,7, 8,9,10,11,12,13,14,15);
}

// B-fragment (32x16 f16, K x N) straight from global weights stored [N, K]
// row-major (so B[k][n] = W[n][k] is K-contiguous per lane):
//   lanes 0-15 : N=lane,    K 0..7,16..23 ;  lanes 16-31 : N=lane-16, K 8..15,24..31
__device__ inline v16h load_bfrag(const f16* __restrict__ W, int K,
                                  int ncol, int lane, int kstep) {
  const int col = ncol + (lane & 15);
  const int kb  = kstep * 32 + ((lane < 16) ? 0 : 8);
  const v8h* p = reinterpret_cast<const v8h*>(W + (size_t)col * K + kb);
  v8h lo = p[0];
  v8h hi = p[2];
  return __builtin_shufflevector(lo, hi, 0,1,2,3,4,5,6,7, 8,9,10,11,12,13,14,15);
}

// ---------------------------------------------------------------------------
__global__ void init_kernel(int* cnt) {
  if (threadIdx.x < N_SP) cnt[threadIdx.x] = 0;
}

__global__ void cvt_kernel(const float* __restrict__ W1,
                           const float* __restrict__ W2,
                           f16* __restrict__ W1h, f16* __restrict__ W2h) {
  const int t  = blockIdx.x * blockDim.x + threadIdx.x;
  const int n1 = N_SP * N_HID * N_IN;
  const int n2 = N_SP * N_HID * N_HID;
  if (t < n1) {
    W1h[t] = (f16)W1[t];
  } else {
    const int i = t - n1;
    if (i < n2) W2h[i] = (f16)W2[i];
  }
}

__global__ void route_kernel(const int* __restrict__ z,
                             const float* __restrict__ E,
                             int* __restrict__ cnt, int* __restrict__ idx,
                             float* __restrict__ out) {
  const int n = blockIdx.x * blockDim.x + threadIdx.x;
  if (n >= N_ATOMS) return;
  out[n] = 0.0f;                       // atoms with no species contribute 0
  const int zz = z[n];
  int s = -1;
#pragma unroll
  for (int j = 0; j < N_SP; ++j)
    if (E[zz * N_SP + j] != 0.0f) s = j;   // one-hot lookup
  if (s >= 0) {
    const int slot = atomicAdd(&cnt[s], 1);
    idx[s * N_ATOMS + slot] = n;
  }
}

// ---------------------------------------------------------------------------
__global__ __launch_bounds__(256) void mlp_kernel(
    const float* __restrict__ x, const int* __restrict__ cnt,
    const int* __restrict__ idx,
    const f16* __restrict__ W1h, const float* __restrict__ b1,
    const f16* __restrict__ W2h, const float* __restrict__ b2,
    const float* __restrict__ W3, const float* __restrict__ b3,
    float* __restrict__ out) {
  const int s     = blockIdx.y;       // species
  const int tile  = blockIdx.x;       // MT-atom tile within bucket
  const int count = cnt[s];
  if (tile * MT >= count) return;     // uniform early exit (empty tile)

  __shared__ __align__(16) f16  sXA[MT * XS];   // 16.5 KB
  __shared__ __align__(16) f16  sH1[MT * HS];   // 33.3 KB
  __shared__ __align__(16) f16  sH2[MT * HS];   // 33.3 KB
  __shared__ float sRed[8 * MT];
  __shared__ int   sAtom[MT];

  const int tid  = threadIdx.x;
  const int lane = tid & 31;
  const int wave = tid >> 5;          // 0..7, each owns 64 hidden columns

  if (tid < MT) {
    const int slot = tile * MT + tid;
    sAtom[tid] = (slot < count) ? idx[s * N_ATOMS + slot]
                                : idx[s * N_ATOMS];   // clamp padding rows
  }
  __syncthreads();

  // Gather MT atoms' x rows, fp32 -> f16 into LDS (32 elems per thread)
  {
    const int row = tid >> 3;          // 0..31
    const int kb  = (tid & 7) * 32;    // 0..224
    const float* xp = x + (size_t)sAtom[row] * N_IN + kb;
    f16* dp = sXA + row * XS + kb;
#pragma unroll
    for (int i = 0; i < 32; ++i) dp[i] = (f16)xp[i];
  }
  __syncthreads();

  // ---------------- Layer 1: [MT x 256] @ W1^T -> tanh -> sH1 [MT x 512]
  {
    v8f acc[2][4] = {{{}, {}, {}, {}}, {{}, {}, {}, {}}};
    const f16* Wb = W1h + (size_t)s * N_HID * N_IN;
    for (int ks = 0; ks < N_IN / 32; ++ks) {
      v16h a0 = load_afrag(sXA, XS, 0,  lane, ks);
      v16h a1 = load_afrag(sXA, XS, 16, lane, ks);
#pragma unroll
      for (int t = 0; t < 4; ++t) {
        const v16h b = load_bfrag(Wb, N_IN, wave * 64 + t * 16, lane, ks);
        acc[0][t] = __builtin_amdgcn_wmma_f32_16x16x32_f16(
            false, a0, false, b, (short)0, acc[0][t], false, false);
        acc[1][t] = __builtin_amdgcn_wmma_f32_16x16x32_f16(
            false, a1, false, b, (short)0, acc[1][t], false, false);
      }
    }
#pragma unroll
    for (int rt = 0; rt < 2; ++rt) {
#pragma unroll
      for (int t = 0; t < 4; ++t) {
        const int ncol   = wave * 64 + t * 16 + (lane & 15);
        const float bias = b1[s * N_HID + ncol];
        const int mbase  = rt * 16 + ((lane < 16) ? 0 : 8);  // C: VGPR r -> M=r / 8+r
#pragma unroll
        for (int r = 0; r < 8; ++r)
          sH1[(mbase + r) * HS + ncol] = (f16)fast_tanh(acc[rt][t][r] + bias);
      }
    }
  }
  __syncthreads();

  // ---------------- Layer 2: [MT x 512] @ W2^T -> tanh -> sH2 [MT x 512]
  {
    v8f acc[2][4] = {{{}, {}, {}, {}}, {{}, {}, {}, {}}};
    const f16* Wb = W2h + (size_t)s * N_HID * N_HID;
    for (int ks = 0; ks < N_HID / 32; ++ks) {
      v16h a0 = load_afrag(sH1, HS, 0,  lane, ks);
      v16h a1 = load_afrag(sH1, HS, 16, lane, ks);
#pragma unroll
      for (int t = 0; t < 4; ++t) {
        const v16h b = load_bfrag(Wb, N_HID, wave * 64 + t * 16, lane, ks);
        acc[0][t] = __builtin_amdgcn_wmma_f32_16x16x32_f16(
            false, a0, false, b, (short)0, acc[0][t], false, false);
        acc[1][t] = __builtin_amdgcn_wmma_f32_16x16x32_f16(
            false, a1, false, b, (short)0, acc[1][t], false, false);
      }
    }
#pragma unroll
    for (int rt = 0; rt < 2; ++rt) {
#pragma unroll
      for (int t = 0; t < 4; ++t) {
        const int ncol   = wave * 64 + t * 16 + (lane & 15);
        const float bias = b2[s * N_HID + ncol];
        const int mbase  = rt * 16 + ((lane < 16) ? 0 : 8);
#pragma unroll
        for (int r = 0; r < 8; ++r)
          sH2[(mbase + r) * HS + ncol] = (f16)fast_tanh(acc[rt][t][r] + bias);
      }
    }
  }
  __syncthreads();

  // ---------------- Layer 3: 512-wide dot with W3[s], fp32 VALU
  {
    const int a = tid & 31;   // atom in tile
    const int c = tid >> 5;   // 0..7, 64-element chunk
    const float* w3 = W3 + s * N_HID;
    float p = 0.0f;
#pragma unroll
    for (int j = 0; j < 64; ++j) {
      const int h = c * 64 + j;
      p += (float)sH2[a * HS + h] * w3[h];
    }
    sRed[c * MT + a] = p;
  }
  __syncthreads();

  if (tid < MT) {
    float sum = b3[s];
#pragma unroll
    for (int c = 0; c < 8; ++c) sum += sRed[c * MT + tid];
    const int slot = tile * MT + tid;
    if (slot < count) out[sAtom[tid]] = sum;
  }
}

// ---------------------------------------------------------------------------
extern "C" void kernel_launch(void* const* d_in, const int* in_sizes, int n_in,
                              void* d_out, int out_size, void* d_ws, size_t ws_size,
                              hipStream_t stream) {
  const float* x  = (const float*)d_in[0];
  const int*   z  = (const int*)  d_in[1];
  const float* W1 = (const float*)d_in[2];
  const float* b1 = (const float*)d_in[3];
  const float* W2 = (const float*)d_in[4];
  const float* b2 = (const float*)d_in[5];
  const float* W3 = (const float*)d_in[6];
  const float* b3 = (const float*)d_in[7];
  const float* E  = (const float*)d_in[8];
  float* out = (float*)d_out;

  char* ws  = (char*)d_ws;
  int*  cnt = (int*)(ws + CNT_OFF);
  int*  idx = (int*)(ws + IDX_OFF);
  f16*  W1h = (f16*)(ws + W1H_OFF);
  f16*  W2h = (f16*)(ws + W2H_OFF);

  init_kernel<<<1, 32, 0, stream>>>(cnt);

  const int nConv = N_SP * N_HID * N_IN + N_SP * N_HID * N_HID;  // 1,572,864
  cvt_kernel<<<(nConv + 255) / 256, 256, 0, stream>>>(W1, W2, W1h, W2h);

  route_kernel<<<N_ATOMS / 256, 256, 0, stream>>>(z, E, cnt, idx, out);

  dim3 grid(N_ATOMS / MT, N_SP);  // worst case: one species owns every atom
  mlp_kernel<<<grid, 256, 0, stream>>>(x, cnt, idx, W1h, b1, W2h, b2, W3, b3, out);
}
